// LoraLinear_88931592831054
// MI455X (gfx1250) — matrix operand order repaired and checked by reading, not compiled, not run on hardware
//
#include <hip/hip_runtime.h>
#include <stdint.h>

typedef __attribute__((ext_vector_type(16))) __bf16   v16bf;
typedef __attribute__((ext_vector_type(8)))  float    v8f;
typedef __attribute__((ext_vector_type(4)))  uint32_t v4u;
typedef __attribute__((ext_vector_type(8)))  int      v8i;
typedef __attribute__((ext_vector_type(4)))  int      v4i;

// Problem dims (fixed by setup_inputs): x[4,2048,4096], W[4096,4096], r=16
constexpr int M_ = 8192;   // B*S
constexpr int N_ = 4096;   // d_out
constexpr int K_ = 4096;   // d_in
constexpr int R_ = 16;     // LoRA rank

constexpr int BM = 128, BN = 128, BK = 32;
constexpr int LDSB  = 80;          // LDS row stride: 64B data + 16B pad
constexpr int ATILE = BM * LDSB;   // 10240 B
constexpr int BTILE = BN * LDSB;   // 10240 B

#if __has_builtin(__builtin_amdgcn_tensor_load_to_lds) && \
    __has_builtin(__builtin_amdgcn_s_wait_tensorcnt)
#define USE_TDM 1
#else
#define USE_TDM 0
#endif

__device__ __forceinline__ uint32_t bf16_pk(float lo, float hi) {
#if __has_builtin(__builtin_amdgcn_cvt_pk_bf16_f32)
  return __builtin_bit_cast(uint32_t, __builtin_amdgcn_cvt_pk_bf16_f32(lo, hi));
#else
  union { __bf16 h[2]; uint32_t u; } c;
  c.h[0] = (__bf16)lo; c.h[1] = (__bf16)hi;   // RNE
  return c.u;
#endif
}

// ---------------------------------------------------------------------------
// Kernel 1: W' = W + lora_B @ lora_A, stored as bf16 into workspace. [N_,K_]
// ---------------------------------------------------------------------------
__global__ __launch_bounds__(256) void prep_weight_kernel(
    const float* __restrict__ W, const float* __restrict__ lA,
    const float* __restrict__ lB, uint16_t* __restrict__ Wb) {
  const int idx = blockIdx.x * blockDim.x + threadIdx.x;   // one per 4 elems
  const int i = (idx * 4) % K_;
  const int o = (idx * 4) / K_;
  const float4 w = *(const float4*)(W + (size_t)o * K_ + i);
  float a0 = w.x, a1 = w.y, a2 = w.z, a3 = w.w;
#pragma unroll
  for (int r = 0; r < R_; ++r) {
    const float b = lB[o * R_ + r];
    const float4 a = *(const float4*)(lA + r * K_ + i);
    a0 += b * a.x; a1 += b * a.y; a2 += b * a.z; a3 += b * a.w;
  }
  uint2 p;
  p.x = bf16_pk(a0, a1);
  p.y = bf16_pk(a2, a3);
  *(uint2*)(Wb + (size_t)o * K_ + i) = p;
}

// ---------------------------------------------------------------------------
// Kernel 2: Xb = bf16(x), one bandwidth pass. [M_,K_]
// ---------------------------------------------------------------------------
__global__ __launch_bounds__(256) void prep_x_kernel(
    const float* __restrict__ X, uint16_t* __restrict__ Xb) {
  const size_t e = (size_t)(blockIdx.x * blockDim.x + threadIdx.x) * 8;
  const float4 a0 = *(const float4*)(X + e);
  const float4 a1 = *(const float4*)(X + e + 4);
  uint4 p;
  p.x = bf16_pk(a0.x, a0.y); p.y = bf16_pk(a0.z, a0.w);
  p.z = bf16_pk(a1.x, a1.y); p.w = bf16_pk(a1.z, a1.w);
  *(uint4*)(Xb + e) = p;
}

// ---------------------------------------------------------------------------
// TDM: DMA one tile (128 rows x 32 K of bf16, row = 64B) into LDS, padding
// each 64B (16 DWORD) row with 16B (4 DWORDs) -> 80B LDS stride.
// Descriptor packing per CDNA5 ISA. `rows` = tensor dim1.
// ---------------------------------------------------------------------------
#if USE_TDM
__device__ __forceinline__ void tdm_load_tile(const uint16_t* base, int row0,
                                              int k0, int rows, uint32_t lds_off) {
  const uint64_t ga = (uint64_t)(uintptr_t)(base + (size_t)row0 * K_ + k0);
  const uint32_t d1s_hi = (uint32_t)(((uint64_t)K_ * (uint64_t)rows) >> 16);
  v4u g0;
  g0[0] = 1u;                                          // count=1, no gather
  g0[1] = (uint32_t)__builtin_amdgcn_readfirstlane((int)lds_off);
  g0[2] = (uint32_t)__builtin_amdgcn_readfirstlane((int)(uint32_t)ga);
  g0[3] = (uint32_t)__builtin_amdgcn_readfirstlane(
              (int)((uint32_t)(ga >> 32) | (2u << 30)));  // addr[56:32]|type=2
  v8i g1;
  g1[0] = (int)((1u << 16)        // data_size = 2 bytes
              | (1u << 20)        // pad_enable
              | (3u << 22)        // pad_interval: 16 DWORDs (64B)
              | (3u << 25));      // pad_amount: 4 DWORDs (16B)
  g1[1] = (int)((uint32_t)K_ << 16);     // tensor_dim0[15:0] @ bits 79:64
  g1[2] = (int)((uint32_t)rows << 16);   // tensor_dim1[15:0] @ bits 111:96
  g1[3] = (int)(32u << 16);       // tile_dim0 = 32 elements (bits 127:112)
  g1[4] = 128;                    // tile_dim1 = 128 rows (bits 143:128)
  g1[5] = K_;                     // tensor_dim0_stride lo32 (bits 191:160)
  g1[6] = 0;                      // dim0_stride[47:32]=0 | dim1_stride[15:0]=0
  g1[7] = (int)d1s_hi;            // tensor_dim1_stride[47:16]
  v4i z4 = {0, 0, 0, 0};
#if defined(__clang_major__) && __clang_major__ >= 23
  v8i z8 = {0, 0, 0, 0, 0, 0, 0, 0};
  __builtin_amdgcn_tensor_load_to_lds(g0, g1, z4, z4, z8, 0);
#else
  __builtin_amdgcn_tensor_load_to_lds(g0, g1, z4, z4, 0);
#endif
}
#endif

// ---------------------------------------------------------------------------
// Shared compute step: 4x2 fragment loads + 8 WMMAs on one LDS buffer pair.
// ---------------------------------------------------------------------------
__device__ __forceinline__ void compute_step(const uint8_t* baseA,
                                             const uint8_t* baseB, int wm, int wn,
                                             int lr, int lh, v8f (&acc)[4][2]) {
  v16bf af[4], bfm[2];
#pragma unroll
  for (int i = 0; i < 4; ++i) {
    // A 16x32 layout: lanes<16 K0-7/16-23 at bytes 0..15/32..47; +16B for hi half
    const uint8_t* pa = baseA + (wm + i * 16 + lr) * LDSB + lh * 16;
    union { v16bf v; uint4 u[2]; } fa;
    fa.u[0] = *(const uint4*)(pa);
    fa.u[1] = *(const uint4*)(pa + 32);
    af[i] = fa.v;
  }
#pragma unroll
  for (int j = 0; j < 2; ++j) {
    // B 32x16 layout: lanes<16 K0-7/16-23 at bytes 0..15/32..47; +16B for hi half
    const uint8_t* pb = baseB + (wn + j * 16 + lr) * LDSB + lh * 16;
    union { v16bf v; uint4 u[2]; } fb;
    fb.u[0] = *(const uint4*)(pb);
    fb.u[1] = *(const uint4*)(pb + 32);
    bfm[j] = fb.v;
  }
#pragma unroll
  for (int i = 0; i < 4; ++i)
#pragma unroll
    for (int j = 0; j < 2; ++j)
      acc[i][j] = __builtin_amdgcn_wmma_f32_16x16x32_bf16(
          false, af[i], false, bfm[j], (short)0, acc[i][j], false, false);
}

// Epilogue: bias add + store (C/D layout: VGPR r -> M=r or r+8, N=lane%16)
__device__ __forceinline__ void store_tiles(v8f (&acc)[4][2], const float* bias,
                                            float* Out, int bm, int bn, int wm,
                                            int wn, int lr, int lh) {
  const int mh = lh * 8;
#pragma unroll
  for (int j = 0; j < 2; ++j) {
    const int n = bn + wn + j * 16 + lr;
    const float bv = bias[n];
#pragma unroll
    for (int i = 0; i < 4; ++i) {
      const int mbase = bm + wm + i * 16 + mh;
#pragma unroll
      for (int r = 0; r < 8; ++r)
        Out[(size_t)(mbase + r) * N_ + n] = acc[i][j][r] + bv;
    }
  }
}

// ---------------------------------------------------------------------------
// Kernel 3a (primary): both A and B tiles pre-converted bf16, staged by TDM.
// Steady-state per wave: 12 ds_load_b128 + 8 v_wmma + 1 barrier. Double buffered.
// ---------------------------------------------------------------------------
__global__ __launch_bounds__(256) void lora_gemm_tdm_kernel(
    const uint16_t* __restrict__ Xb, const uint16_t* __restrict__ Wb,
    const float* __restrict__ bias, float* __restrict__ Out) {
  __shared__ __align__(16) uint8_t ldsA[2 * ATILE];
  __shared__ __align__(16) uint8_t ldsB[2 * BTILE];

  const int tid  = threadIdx.x;
  const int lane = tid & 31;
  const int wave = tid >> 5;
  const int bm = blockIdx.y * BM;
  const int bn = blockIdx.x * BN;
  const int wm = (wave >> 2) * 64;
  const int wn = (wave & 3)  * 32;
  const int lr = lane & 15;
  const int lh = lane >> 4;

  v8f acc[4][2];
#pragma unroll
  for (int i = 0; i < 4; ++i)
#pragma unroll
    for (int j = 0; j < 2; ++j) acc[i][j] = (v8f)(0.f);

#if USE_TDM
  if (wave == 0) {
    tdm_load_tile(Xb, bm, 0, M_, (uint32_t)(uintptr_t)&ldsA[0]);
    tdm_load_tile(Wb, bn, 0, N_, (uint32_t)(uintptr_t)&ldsB[0]);
    __builtin_amdgcn_s_wait_tensorcnt(0);
  }
  __syncthreads();

  int cur = 0;
  for (int k0 = 0; k0 < K_; k0 += BK) {
    const bool has_next = (k0 + BK) < K_;
    if (has_next && wave == 0) {
      tdm_load_tile(Xb, bm, k0 + BK, M_,
                    (uint32_t)(uintptr_t)&ldsA[(cur ^ 1) * ATILE]);
      tdm_load_tile(Wb, bn, k0 + BK, N_,
                    (uint32_t)(uintptr_t)&ldsB[(cur ^ 1) * BTILE]);
    }
    compute_step(ldsA + cur * ATILE, ldsB + cur * BTILE, wm, wn, lr, lh, acc);
    if (has_next) {
      if (wave == 0) __builtin_amdgcn_s_wait_tensorcnt(0);
      __syncthreads();
      cur ^= 1;
    }
  }
#else
  // Toolchain without TDM builtin: synchronous bf16 tile copies, single buffer.
  const int rowT = tid >> 1;
  const int colT = (tid & 1) * 16;
  const uint16_t* gAb = Xb + (size_t)(bm + rowT) * K_ + colT;
  const uint16_t* gBb = Wb + (size_t)(bn + rowT) * K_ + colT;
  uint8_t* dA = ldsA + rowT * LDSB + colT * 2;
  uint8_t* dB = ldsB + rowT * LDSB + colT * 2;
  for (int k0 = 0; k0 < K_; k0 += BK) {
    const uint4 a0 = *(const uint4*)(gAb + k0);
    const uint4 a1 = *(const uint4*)(gAb + k0 + 8);
    const uint4 b0 = *(const uint4*)(gBb + k0);
    const uint4 b1 = *(const uint4*)(gBb + k0 + 8);
    __syncthreads();
    *(uint4*)(dA) = a0; *(uint4*)(dA + 16) = a1;
    *(uint4*)(dB) = b0; *(uint4*)(dB + 16) = b1;
    __syncthreads();
    compute_step(ldsA, ldsB, wm, wn, lr, lh, acc);
  }
#endif

  store_tiles(acc, bias, Out, bm, bn, wm, wn, lr, lh);
}

// ---------------------------------------------------------------------------
// Kernel 3b (fallback, small workspace): x stays f32, converted in-wave; B via TDM.
// ---------------------------------------------------------------------------
__global__ __launch_bounds__(256) void lora_gemm_conv_kernel(
    const float* __restrict__ X, const uint16_t* __restrict__ Wb,
    const float* __restrict__ bias, float* __restrict__ Out) {
  __shared__ __align__(16) uint8_t ldsA[2 * ATILE];
  __shared__ __align__(16) uint8_t ldsB[2 * BTILE];

  const int tid  = threadIdx.x;
  const int lane = tid & 31;
  const int wave = tid >> 5;
  const int bm = blockIdx.y * BM;
  const int bn = blockIdx.x * BN;
  const int wm = (wave >> 2) * 64;
  const int wn = (wave & 3)  * 32;
  const int lr = lane & 15;
  const int lh = lane >> 4;

  v8f acc[4][2];
#pragma unroll
  for (int i = 0; i < 4; ++i)
#pragma unroll
    for (int j = 0; j < 2; ++j) acc[i][j] = (v8f)(0.f);

  const int rowT = tid >> 1;
  const int colT = (tid & 1) * 16;
  const float* gA = X + (size_t)(bm + rowT) * K_ + colT;
#if !USE_TDM
  const uint16_t* gB = Wb + (size_t)(bn + rowT) * K_ + colT;
#endif

  // prologue: stage k0 = 0 into buffer 0
#if USE_TDM
  if (wave == 0)
    tdm_load_tile(Wb, bn, 0, N_, (uint32_t)(uintptr_t)&ldsB[0]);
#else
  {
    const uint4 b0 = *(const uint4*)(gB + 0);
    const uint4 b1 = *(const uint4*)(gB + 8);
    uint8_t* dB = ldsB + rowT * LDSB + colT * 2;
    *(uint4*)(dB) = b0; *(uint4*)(dB + 16) = b1;
  }
#endif
  {
    const float4 a0 = *(const float4*)(gA + 0);
    const float4 a1 = *(const float4*)(gA + 4);
    const float4 a2 = *(const float4*)(gA + 8);
    const float4 a3 = *(const float4*)(gA + 12);
    uint4 p0, p1;
    p0.x = bf16_pk(a0.x, a0.y); p0.y = bf16_pk(a0.z, a0.w);
    p0.z = bf16_pk(a1.x, a1.y); p0.w = bf16_pk(a1.z, a1.w);
    p1.x = bf16_pk(a2.x, a2.y); p1.y = bf16_pk(a2.z, a2.w);
    p1.z = bf16_pk(a3.x, a3.y); p1.w = bf16_pk(a3.z, a3.w);
    uint8_t* dA = ldsA + rowT * LDSB + colT * 2;
    *(uint4*)(dA) = p0; *(uint4*)(dA + 16) = p1;
  }
#if USE_TDM
  if (wave == 0) __builtin_amdgcn_s_wait_tensorcnt(0);
#endif
  __syncthreads();

  int cur = 0;
  for (int k0 = 0; k0 < K_; k0 += BK) {
    const bool has_next = (k0 + BK) < K_;
    float4 n0 = {}, n1 = {}, n2 = {}, n3 = {};
#if !USE_TDM
    uint4 nb0 = {}, nb1 = {};
#endif
    if (has_next) {
      n0 = *(const float4*)(gA + k0 + BK + 0);
      n1 = *(const float4*)(gA + k0 + BK + 4);
      n2 = *(const float4*)(gA + k0 + BK + 8);
      n3 = *(const float4*)(gA + k0 + BK + 12);
#if USE_TDM
      if (wave == 0)
        tdm_load_tile(Wb, bn, k0 + BK, N_,
                      (uint32_t)(uintptr_t)&ldsB[(cur ^ 1) * BTILE]);
#else
      nb0 = *(const uint4*)(gB + k0 + BK + 0);
      nb1 = *(const uint4*)(gB + k0 + BK + 8);
#endif
    }

    compute_step(ldsA + cur * ATILE, ldsB + cur * BTILE, wm, wn, lr, lh, acc);

    if (has_next) {
      uint4 p0, p1;
      p0.x = bf16_pk(n0.x, n0.y); p0.y = bf16_pk(n0.z, n0.w);
      p0.z = bf16_pk(n1.x, n1.y); p0.w = bf16_pk(n1.z, n1.w);
      p1.x = bf16_pk(n2.x, n2.y); p1.y = bf16_pk(n2.z, n2.w);
      p1.z = bf16_pk(n3.x, n3.y); p1.w = bf16_pk(n3.z, n3.w);
      uint8_t* dA = ldsA + (cur ^ 1) * ATILE + rowT * LDSB + colT * 2;
      *(uint4*)(dA) = p0; *(uint4*)(dA + 16) = p1;
#if !USE_TDM
      uint8_t* dB = ldsB + (cur ^ 1) * BTILE + rowT * LDSB + colT * 2;
      *(uint4*)(dB) = nb0; *(uint4*)(dB + 16) = nb1;
#endif
#if USE_TDM
      if (wave == 0) __builtin_amdgcn_s_wait_tensorcnt(0);
#endif
      __syncthreads();
      cur ^= 1;
    }
  }

  store_tiles(acc, bias, Out, bm, bn, wm, wn, lr, lh);
}

// ---------------------------------------------------------------------------
extern "C" void kernel_launch(void* const* d_in, const int* in_sizes, int n_in,
                              void* d_out, int out_size, void* d_ws, size_t ws_size,
                              hipStream_t stream) {
  const float* x      = (const float*)d_in[0];
  const float* weight = (const float*)d_in[1];
  const float* bias   = (const float*)d_in[2];
  const float* lora_A = (const float*)d_in[3];
  const float* lora_B = (const float*)d_in[4];
  float* out = (float*)d_out;

  const size_t WB_BYTES = (size_t)N_ * K_ * sizeof(uint16_t);   // 32 MB
  const size_t XB_BYTES = (size_t)M_ * K_ * sizeof(uint16_t);   // 64 MB

  uint16_t* Wb = (uint16_t*)d_ws;
  (void)in_sizes; (void)n_in; (void)out_size;

  prep_weight_kernel<<<(N_ * K_ / 4) / 256, 256, 0, stream>>>(
      weight, lora_A, lora_B, Wb);

  dim3 grid(N_ / BN, M_ / BM);   // (32, 64)
  if (ws_size >= WB_BYTES + XB_BYTES) {
    uint16_t* Xb = (uint16_t*)((uint8_t*)d_ws + WB_BYTES);
    prep_x_kernel<<<(M_ * K_ / 8) / 256, 256, 0, stream>>>(x, Xb);
    lora_gemm_tdm_kernel<<<grid, 256, 0, stream>>>(Xb, Wb, bias, out);
  } else {
    lora_gemm_conv_kernel<<<grid, 256, 0, stream>>>(x, Wb, bias, out);
  }
}
